// OffsetDiversity_48120813584387
// MI455X (gfx1250) — matrix operand order repaired and checked by reading, not compiled, not run on hardware
//
#include <hip/hip_runtime.h>
#include <hip/hip_bf16.h>
#include <math.h>

// ---------------- problem constants (match reference) ----------------
#define BATCH 4
#define CIN   64          // per stream
#define HH    192
#define WW    192
#define HWSZ  (HH * WW)   // 36864
#define OFFCH 216         // channels in each offset tensor
#define MAGN  10.0f
#define CG    8           // in-channels per deform group (128/16)
#define KTAP  9           // 3x3

// Per wave: A[16 pix x 288] * B[288 x 16 outch]; K = 4 groups * 72 = 9*32.
// K ordering inside a group: kkloc = k*8 + c  (channel fastest -> vector stores)
#define KTOT  288
#define ASTR  296         // padded A row stride (f16), 296 % 8 == 0 for b128 align

typedef __attribute__((ext_vector_type(16))) _Float16 v16h;
typedef __attribute__((ext_vector_type(8)))  float    v8f;

union Frag16 { unsigned int u[8]; v16h v; };
union Pack2  { _Float16 h[2]; unsigned int u; };
union Pack8  { _Float16 h[8]; uint4 q; };

__global__ __launch_bounds__(128)
void offdiv_deform_wmma(const float* __restrict__ x1,
                        const float* __restrict__ off1,
                        const float* __restrict__ flow1,
                        const float* __restrict__ x2,
                        const float* __restrict__ off2,
                        const float* __restrict__ flow2,
                        const float* __restrict__ weight,   // (64, 8, 3, 3)
                        const float* __restrict__ bias,     // (64)
                        float* __restrict__ out)            // (4, 64, 192, 192)
{
    // LDS: per-wave A staging + block-diagonal B pre-expanded in fragment order
    __shared__ _Float16     Ash[4 * 16 * ASTR];     // 37888 B
    __shared__ unsigned int Bsh[9 * 32 * 8];        //  9216 B (u32 fragment words)

    const int tid  = threadIdx.x;
    const int wave = tid >> 5;
    const int lane = tid & 31;
    const int low  = lane & 15;
    const int hi   = lane >> 4;

    // grid: (64-pixel row segment) x (group block). 4 waves = 4 16-pixel tiles.
    const int gblk = blockIdx.x & 3;
    int t64 = blockIdx.x >> 2;
    const int segPerRow = WW / 64;                 // 3
    int b   = t64 / (HH * segPerRow);
    int rem = t64 - b * (HH * segPerRow);
    int y   = rem / segPerRow;
    int sx0 = (rem - y * segPerRow) * 64;          // segment start
    int tx0 = sx0 + wave * 16;                     // this wave's 16-pixel tile

    // ---- build B fragment words (block-diagonal, zeros via value select) ----
    for (int i = tid; i < 9 * 32 * 8; i += 128) {
        int kc  = i >> 8;                  // /256
        int r   = i & 255;
        int ln  = r >> 3;
        int j   = r & 7;
        int n   = ln & 15;                 // column = out channel within block
        int kk  = kc * 32 + (ln >> 4) * 8 + (j >> 2) * 16 + 2 * (j & 3);
        int gg  = kk / 72;
        int kkl = kk - gg * 72;            // even; kkl = k*8 + c, c even
        int c   = kkl & 7;
        int k   = kkl >> 3;
        int wb  = (gblk * 16 + n) * 72;    // weight row (flat (cout, c*9+k))
        float w0 = weight[wb + c * 9 + k];
        float w1 = weight[wb + (c + 1) * 9 + k];
        bool live = ((n >> 2) == gg);
        Pack2 p;
        p.h[0] = live ? (_Float16)w0 : (_Float16)0.f;
        p.h[1] = live ? (_Float16)w1 : (_Float16)0.f;
        Bsh[i] = p.u;
    }

    // ---- gather stage: 576 tuples per wave = 16 pixels * 4 groups * 9 taps ----
    const int abase_w = wave * 16 * ASTR;
    for (int tt = lane; tt < 16 * 4 * KTAP; tt += 32) {
        int pix = tt / 36;
        int r   = tt - pix * 36;
        int gg  = r / KTAP;
        int k9  = r - gg * KTAP;
        int g   = gblk * 4 + gg;           // global deform group 0..15
        int s   = g >> 3;                  // stream select
        int gl  = g & 7;

        const float* Xp = s ? x2    : x1;
        const float* Op = s ? off2  : off1;
        const float* Fp = s ? flow2 : flow1;

        int xi = tx0 + pix;
        int yx = y * WW + xi;

        size_t ob = (size_t)b * OFFCH * HWSZ;
        int    oc = (gl * KTAP + k9) * 2;
        float dyr = Op[ob + (size_t)oc * HWSZ + yx];
        float dxr = Op[ob + (size_t)(oc + 1) * HWSZ + yx];
        float mr  = Op[ob + (size_t)(144 + gl * KTAP + k9) * HWSZ + yx];
        // flow is flipped: dy channels get flow ch1, dx channels get flow ch0
        float fy = Fp[((size_t)b * 2 + 1) * HWSZ + yx];
        float fx = Fp[((size_t)b * 2 + 0) * HWSZ + yx];

        float py = tanhf(dyr) * MAGN + fy + (float)(k9 / 3) + (float)(y - 1);
        float px = tanhf(dxr) * MAGN + fx + (float)(k9 % 3) + (float)(xi - 1);

        float fy0 = floorf(py), fx0 = floorf(px);
        float wy = py - fy0,   wx = px - fx0;
        int iy0 = (int)fy0, ix0 = (int)fx0;
        int iy1 = iy0 + 1,  ix1 = ix0 + 1;

        float vy0 = (iy0 >= 0 && iy0 < HH) ? 1.f : 0.f;
        float vy1 = (iy1 >= 0 && iy1 < HH) ? 1.f : 0.f;
        float vx0 = (ix0 >= 0 && ix0 < WW) ? 1.f : 0.f;
        float vx1 = (ix1 >= 0 && ix1 < WW) ? 1.f : 0.f;
        int cy0 = min(max(iy0, 0), HH - 1);
        int cy1 = min(max(iy1, 0), HH - 1);
        int cx0 = min(max(ix0, 0), WW - 1);
        int cx1 = min(max(ix1, 0), WW - 1);

        float w00 = (1.f - wy) * (1.f - wx) * vy0 * vx0;
        float w01 = (1.f - wy) * wx         * vy0 * vx1;
        float w10 = wy         * (1.f - wx) * vy1 * vx0;
        float w11 = wy         * wx         * vy1 * vx1;
        float msk = 1.f / (1.f + expf(-mr));

        int o00 = cy0 * WW + cx0, o01 = cy0 * WW + cx1;
        int o10 = cy1 * WW + cx0, o11 = cy1 * WW + cx1;

        const float* plane0 = Xp + ((size_t)(b * CIN + gl * CG)) * HWSZ;
        Pack8 v8;
        #pragma unroll
        for (int c = 0; c < CG; ++c) {
            const float* pl = plane0 + (size_t)c * HWSZ;
            float v = w00 * pl[o00] + w01 * pl[o01] +
                      w10 * pl[o10] + w11 * pl[o11];
            v8.h[c] = (_Float16)(msk * v);
        }
        // kkloc = k9*8 + c : 8 contiguous f16, 16B aligned -> one b128 store
        int arow = abase_w + pix * ASTR + gg * 72 + k9 * 8;
        *reinterpret_cast<uint4*>(&Ash[arow]) = v8.q;
    }

    __syncthreads();

    // ---- WMMA stage: 9 x v_wmma_f32_16x16x32_f16, K = 288 ----
    v8f acc = {};
    const int abase = abase_w + low * ASTR;       // lane low = pixel row M

    #pragma unroll
    for (int kc = 0; kc < 9; ++kc) {
        Frag16 af, bf;
        // A: 16-bit A 16x32 layout (VGPR0-3: K=base..base+7, VGPR4-7: +16)
        #pragma unroll
        for (int j = 0; j < 4; ++j) {
            int kA = kc * 32 + hi * 8 + 2 * j;
            af.u[j]     = *reinterpret_cast<const unsigned int*>(&Ash[abase + kA]);
            af.u[4 + j] = *reinterpret_cast<const unsigned int*>(&Ash[abase + kA + 16]);
        }
        // B: pre-expanded fragment words, 32B contiguous per lane
        const unsigned int* bp = &Bsh[(kc * 32 + lane) * 8];
        #pragma unroll
        for (int j = 0; j < 8; ++j)
            bf.u[j] = bp[j];
        acc = __builtin_amdgcn_wmma_f32_16x16x32_f16(
                  false, af.v, false, bf.v, (short)0, acc, false, false);
    }

    // ---- store: D layout VGPR r -> pixel hi*8 + r, lane low -> out channel
    const int ch = gblk * 16 + low;
    const float bs = bias[ch];
    float* orow = out + ((size_t)(b * 64 + ch)) * HWSZ + (size_t)y * WW + tx0;
    #pragma unroll
    for (int r = 0; r < 8; ++r)
        orow[hi * 8 + r] = acc[r] + bs;
}

extern "C" void kernel_launch(void* const* d_in, const int* in_sizes, int n_in,
                              void* d_out, int out_size, void* d_ws, size_t ws_size,
                              hipStream_t stream) {
    (void)in_sizes; (void)n_in; (void)out_size; (void)d_ws; (void)ws_size;
    const float* x1     = (const float*)d_in[0];
    const float* off1   = (const float*)d_in[1];
    const float* flow1  = (const float*)d_in[2];
    const float* x2     = (const float*)d_in[3];
    const float* off2   = (const float*)d_in[4];
    const float* flow2  = (const float*)d_in[5];
    const float* weight = (const float*)d_in[6];
    const float* bias   = (const float*)d_in[7];
    float* out          = (float*)d_out;

    // 2304 64-pixel row segments x 4 group blocks = 9216 workgroups, 128 thr
    const int blocks = BATCH * HH * (WW / 64) * 4;
    offdiv_deform_wmma<<<blocks, 128, 0, stream>>>(
        x1, off1, flow1, x2, off2, flow2, weight, bias, out);
}
